// Model_29222957481995
// MI455X (gfx1250) — matrix operand order repaired and checked by reading, not compiled, not run on hardware
//
#include <hip/hip_runtime.h>
#include <math.h>

#define BATCH     128
#define SEQ_LEN   720
#define PRED_LEN  96
#define CHANNELS  862
#define HIDDEN    16
#define IN_LEN    360
#define KPAD      384          // K padded to multiple of 32
#define NCHUNK    (KPAD / 32)  // 12

typedef __attribute__((ext_vector_type(16))) __bf16 v16bf;
typedef __attribute__((ext_vector_type(8)))  float  v8f;
typedef __attribute__((ext_vector_type(4)))  unsigned int v4u;
typedef __attribute__((ext_vector_type(8)))  int    v8i;
typedef __attribute__((ext_vector_type(4)))  int    v4i;

__device__ __forceinline__ unsigned short f32_to_bf16(float f) {
    union { float f; unsigned int u; } c; c.f = f;
    unsigned int u = c.u;
    unsigned int r = u + 0x7FFFu + ((u >> 16) & 1u);   // round to nearest even
    return (unsigned short)(r >> 16);
}

// ---------------------------------------------------------------------------
// prep 1: A1[h][n] = (1/sqrt(360)) * sum_k w1[h][k] * D[k][n]   (n zero-padded)
// D[k][n] = s_k cos(pi*(n+0.5)*k/360), s_0=sqrt(1/360), s_k=sqrt(2/360)
// ---------------------------------------------------------------------------
__global__ void prep_a1(const float* __restrict__ w1, float* __restrict__ a1) {
    int idx = blockIdx.x * blockDim.x + threadIdx.x;
    if (idx >= HIDDEN * KPAD) return;
    int h = idx / KPAD, n = idx % KPAD;
    float acc = 0.0f;
    if (n < IN_LEN) {
        const float PI   = 3.14159265358979323846f;
        const float invN = 1.0f / (float)IN_LEN;
        const float s0   = sqrtf(1.0f / (float)IN_LEN);
        const float sk   = sqrtf(2.0f / (float)IN_LEN);
        const float ang0 = PI * ((float)n + 0.5f) * invN;
        for (int k = 0; k < IN_LEN; ++k) {
            float d = cosf(ang0 * (float)k) * ((k == 0) ? s0 : sk);
            acc += w1[h * IN_LEN + k] * d;
        }
        acc *= rsqrtf((float)IN_LEN);  // extra 1/sqrt(n) from reference
    }
    a1[h * KPAD + n] = acc;
}

// ---------------------------------------------------------------------------
// prep 2: G[p][n] = sum_h w2[p][h]*A1[h][n];  S = sum_n G[p][n]
// F[p][n] = inv_sqrt2*G + (1 - sqrt2*S)/720   (mean term folded in), bf16
// bias[p] = sum_h b1[h]*w2[p][h] + b2[p]
// ---------------------------------------------------------------------------
__global__ void prep_f(const float* __restrict__ w2, const float* __restrict__ b1,
                       const float* __restrict__ b2, const float* __restrict__ a1,
                       unsigned short* __restrict__ Fmat, float* __restrict__ bias) {
    __shared__ float red[KPAD];
    const int p = blockIdx.x;    // 0..95
    const int n = threadIdx.x;   // 0..383
    float g = 0.0f;
    #pragma unroll
    for (int h = 0; h < HIDDEN; ++h)
        g += w2[p * HIDDEN + h] * a1[h * KPAD + n];   // a1 zero-padded for n>=360
    red[n] = g;
    __syncthreads();
    if (n < 128) red[n] += red[n + 256];
    __syncthreads();
    for (int s = 128; s > 0; s >>= 1) {
        if (n < s) red[n] += red[n + s];
        __syncthreads();
    }
    const float S = red[0];
    const float INV_SQRT2 = 0.70710678118654752440f;
    const float SQRT2     = 1.41421356237309504880f;
    float fv = 0.0f;
    if (n < IN_LEN) fv = INV_SQRT2 * g + (1.0f - SQRT2 * S) / (float)SEQ_LEN;
    Fmat[p * KPAD + n] = f32_to_bf16(fv);
    if (n == 0) {
        float bb = b2[p];
        #pragma unroll
        for (int h = 0; h < HIDDEN; ++h) bb += b1[h] * w2[p * HIDDEN + h];
        bias[p] = bb;
    }
}

// ---------------------------------------------------------------------------
// main: out[b][p][c] = sum_n s2[b,c,n]*F[p][n] + bias[p]
// block = 256 threads (8 wave32); block tile: 128 rows (c) x 96 cols (p)
// each wave: one 16(c) x 96(p) strip, K loop of 12 x 32 via wmma bf16.
// F chunk staged global->LDS by the Tensor Data Mover (TDM) with hardware
// row padding (16 dwords + 1 pad dword -> conflict-free 17-dword pitch).
// ---------------------------------------------------------------------------
__global__ __launch_bounds__(256)
void fits_main(const float* __restrict__ x, const unsigned short* __restrict__ Fmat,
               const float* __restrict__ bias, float* __restrict__ out) {
    __shared__ union {
        struct {
            unsigned short s2[128 * 34];  // [c][n] bf16, pitch 34 halves (17 dwords)
            unsigned short fm[96 * 34];   // [p][n] bf16, pitch 34 halves (TDM-padded)
        } st;
        float ob[PRED_LEN * 129];         // [p][c] f32, pitch 129 -> no bank conflicts
    } sm;

    const int b    = blockIdx.y;
    const int c0   = blockIdx.x * 128;
    const int t    = threadIdx.x;
    const int wave = t >> 5;
    const int lane = t & 31;
    const int hi   = (lane >= 16) ? 1 : 0;
    const int l16  = lane & 15;

    v8f acc[6] = {};   // 6 N-tiles of 16 -> 96 columns

    const int cc    = t & 127;                                    // c within block tile
    const int jp    = t >> 7;                                     // n phase 0/1
    const int cload = (c0 + cc < CHANNELS) ? (c0 + cc) : (CHANNELS - 1);
    const float* xb = x + (size_t)b * SEQ_LEN * CHANNELS;

    // TDM descriptor pieces that are loop-invariant
    const unsigned int ldsF = (unsigned int)(uintptr_t)(&sm.st.fm[0]); // LDS aperture offset
    const unsigned long long gbase = (unsigned long long)(uintptr_t)Fmat;
    // group1: data_size=2B (bits17:16=1), pad_enable (bit20), pad_interval=3 (16 dw),
    // pad_amount=0 (1 dw); tensor_dim0=32, tensor_dim1=96, tile_dim0=32, tile_dim1=96,
    // tensor_dim0_stride=384 elements
    const v8i g1 = { (int)0x00D10000, (int)(32u << 16), (int)(96u << 16),
                     (int)(32u << 16), 96, 384, 0, 0 };
    const v4i gz4 = { 0, 0, 0, 0 };
    const v8i gz8 = { 0, 0, 0, 0, 0, 0, 0, 0 };

    for (int kc = 0; kc < NCHUNK; ++kc) {
        const int n0 = kc * 32;

        // ---- kick off F chunk DMA: global [96 x 64B, stride 768B] -> LDS (padded)
        if (wave == 0) {
            const unsigned long long ga = gbase + (unsigned long long)kc * 64ull;
            v4u g0 = { 1u,                        // count=1 valid descriptor
                       ldsF,                      // lds_addr
                       (unsigned int)ga,          // global_addr[31:0]
                       ((unsigned int)(ga >> 32) & 0x01FFFFFFu) | 0x80000000u }; // addr[56:32] | type=2
            __builtin_amdgcn_tensor_load_to_lds(g0, g1, gz4, gz4, gz8, 0);
        }

        // ---- stage s2 chunk (bf16): coalesced 512B row segments, read-once -> NT
        for (int j = jp; j < 32; j += 2) {
            const int n   = n0 + j;
            const int row = 2 * n;
            float v = 0.0f;
            if (row < SEQ_LEN) {
                const float* rp = xb + (size_t)row * CHANNELS + cload;
                float e0 = __builtin_nontemporal_load(rp);
                float e1 = __builtin_nontemporal_load(rp + CHANNELS);
                v = e0 + e1;
            }
            sm.st.s2[cc * 34 + j] = f32_to_bf16(v);
        }

        if (wave == 0) __builtin_amdgcn_s_wait_tensorcnt(0);
        __syncthreads();

        // ---- A fragment: 16(c) x 32(n) bf16, CDNA5 A layout
        union { v16bf v; unsigned int u[8]; } afrag;
        {
            const unsigned int* S = (const unsigned int*)sm.st.s2;
            const int crow = wave * 16 + l16;
            #pragma unroll
            for (int r = 0; r < 8; ++r) {
                const int kb = ((r < 4) ? 0 : 16) + (hi ? 8 : 0) + 2 * (r & 3);
                afrag.u[r] = S[crow * 17 + (kb >> 1)];
            }
        }

        // ---- B fragments (32(n) x 16(p)) + WMMA
        const unsigned int* FB = (const unsigned int*)sm.st.fm;
        #pragma unroll
        for (int pt = 0; pt < 6; ++pt) {
            union { v16bf v; unsigned int u[8]; } bfrag;
            const int prow = pt * 16 + l16;
            #pragma unroll
            for (int r = 0; r < 8; ++r)
                bfrag.u[r] = FB[prow * 17 + ((r < 4) ? 0 : 8) + (hi ? 4 : 0) + (r & 3)];
            acc[pt] = __builtin_amdgcn_wmma_f32_16x16x32_bf16(
                false, afrag.v, false, bfrag.v, (short)0, acc[pt], false, false);
        }
        __syncthreads();   // fm/s2 consumed; safe to restage next chunk
    }

    // ---- epilogue: D tiles -> LDS [p][c], then coalesced NT stores + bias
    #pragma unroll
    for (int pt = 0; pt < 6; ++pt) {
        #pragma unroll
        for (int r = 0; r < 8; ++r) {
            const int p  = pt * 16 + l16;
            const int cl = wave * 16 + r + (hi ? 8 : 0);
            sm.ob[p * 129 + cl] = acc[pt][r];
        }
    }
    __syncthreads();

    float* ob = out + (size_t)b * PRED_LEN * CHANNELS;
    for (int i = t; i < PRED_LEN * 128; i += 256) {
        const int p  = i >> 7;
        const int ci = i & 127;
        const int c  = c0 + ci;
        if (c < CHANNELS) {
            float v = sm.ob[p * 129 + ci] + bias[p];
            __builtin_nontemporal_store(v, ob + (size_t)p * CHANNELS + c);
        }
    }
}

// ---------------------------------------------------------------------------
extern "C" void kernel_launch(void* const* d_in, const int* in_sizes, int n_in,
                              void* d_out, int out_size, void* d_ws, size_t ws_size,
                              hipStream_t stream) {
    (void)in_sizes; (void)n_in; (void)out_size; (void)ws_size;
    const float* x  = (const float*)d_in[0];
    const float* w1 = (const float*)d_in[1];
    const float* b1 = (const float*)d_in[2];
    const float* w2 = (const float*)d_in[3];
    const float* b2 = (const float*)d_in[4];
    float* out = (float*)d_out;

    // workspace layout: F (96x384 bf16) | A1 (16x384 f32) | bias (96 f32)
    unsigned short* Fmat = (unsigned short*)d_ws;
    float* a1   = (float*)((char*)d_ws + (size_t)PRED_LEN * KPAD * 2);
    float* bias = (float*)((char*)d_ws + (size_t)PRED_LEN * KPAD * 2 + (size_t)HIDDEN * KPAD * 4);

    prep_a1<<<(HIDDEN * KPAD + 255) / 256, 256, 0, stream>>>(w1, a1);
    prep_f<<<PRED_LEN, KPAD, 0, stream>>>(w2, b1, b2, a1, Fmat, bias);

    dim3 grid((CHANNELS + 127) / 128, BATCH);
    fits_main<<<grid, 256, 0, stream>>>(x, Fmat, bias, out);
}
